// SparseAutoencoder_87900800680644
// MI455X (gfx1250) — matrix operand order repaired and checked by reading, not compiled, not run on hardware
//
#include <hip/hip_runtime.h>
#include <hip/hip_bf16.h>

#define BATCH   2048
#define IN_DIM  1024
#define HID_DIM 32768
#define TOPK    32
#define LDSK    40   // padded LDS row stride in bf16 elems (80B = 5*16B, keeps b128 aligned)

typedef __attribute__((ext_vector_type(16))) __bf16 v16bf;
typedef __attribute__((ext_vector_type(8)))  float  v8f;

union FragAB { v16bf v; uint4 u[2]; };

__device__ __forceinline__ unsigned int f2bf2(float lo, float hi) {
  // round-to-nearest-even fp32 -> bf16, packed pair
  unsigned a = __float_as_uint(lo); a = a + 0x7FFFu + ((a >> 16) & 1u);
  unsigned b = __float_as_uint(hi); b = b + 0x7FFFu + ((b >> 16) & 1u);
  return (a >> 16) | (b & 0xFFFF0000u);
}

// ---------------- Kernel 1: encode GEMM (x @ W_enc^T + b_enc, ReLU) ----------------
__global__ __launch_bounds__(256)
void sae_encode_wmma(const float* __restrict__ x,
                     const float* __restrict__ Wenc,
                     const float* __restrict__ bEnc,
                     float* __restrict__ h)
{
  __shared__ unsigned short sX[128 * LDSK];
  __shared__ unsigned short sW[128 * LDSK];

  const int tid  = threadIdx.x;
  const int lane = tid & 31;
  const int wave = tid >> 5;
  const int wm   = (wave & 1) * 64;   // wave M offset inside 128-tile
  const int wn   = (wave >> 1) * 32;  // wave N offset inside 128-tile
  const int l16  = lane & 15;
  const int half = lane >> 4;

  v8f acc[4][2];
  #pragma unroll
  for (int mi = 0; mi < 4; ++mi)
    #pragma unroll
    for (int ni = 0; ni < 2; ++ni)
      acc[mi][ni] = (v8f){0.f,0.f,0.f,0.f,0.f,0.f,0.f,0.f};

  const float* xBase = x    + (size_t)blockIdx.y * 128 * IN_DIM;
  const float* wBase = Wenc + (size_t)blockIdx.x * 128 * IN_DIM;

  for (int kt = 0; kt < IN_DIM; kt += 32) {
    // stage fp32 tiles -> bf16 LDS (128 rows x 32 K each)
    #pragma unroll
    for (int i = 0; i < 4; ++i) {
      int f   = tid + i * 256;
      int row = f >> 3;
      int ch  = f & 7;
      float4 qx = *(const float4*)(xBase + (size_t)row * IN_DIM + kt + ch * 4);
      float4 qw = *(const float4*)(wBase + (size_t)row * IN_DIM + kt + ch * 4);
      uint2 px; px.x = f2bf2(qx.x, qx.y); px.y = f2bf2(qx.z, qx.w);
      uint2 pw; pw.x = f2bf2(qw.x, qw.y); pw.y = f2bf2(qw.z, qw.w);
      *(uint2*)&sX[row * LDSK + ch * 4] = px;
      *(uint2*)&sW[row * LDSK + ch * 4] = pw;
    }
    __syncthreads();

    FragAB A[4], B[2];
    #pragma unroll
    for (int mi = 0; mi < 4; ++mi) {
      // A 16x32 bf16: lane group selects K chunks {8*half .. +8, +16 .. +24}
      const uint4* p = (const uint4*)&sX[(wm + mi * 16 + l16) * LDSK + half * 8];
      A[mi].u[0] = p[0]; A[mi].u[1] = p[2];
    }
    #pragma unroll
    for (int ni = 0; ni < 2; ++ni) {
      // B 32x16 bf16: lane group selects K chunks {8*half .. +8, +16 .. +24}
      const uint4* p = (const uint4*)&sW[(wn + ni * 16 + l16) * LDSK + half * 8];
      B[ni].u[0] = p[0]; B[ni].u[1] = p[2];
    }
    #pragma unroll
    for (int mi = 0; mi < 4; ++mi)
      #pragma unroll
      for (int ni = 0; ni < 2; ++ni)
        acc[mi][ni] = __builtin_amdgcn_wmma_f32_16x16x32_bf16(
            false, A[mi].v, false, B[ni].v, (short)0, acc[mi][ni], false, false);
    __syncthreads();
  }

  // epilogue: + b_enc, ReLU, streaming (nontemporal) store of h
  #pragma unroll
  for (int ni = 0; ni < 2; ++ni) {
    int n    = blockIdx.x * 128 + wn + ni * 16 + l16;
    float be = bEnc[n];
    #pragma unroll
    for (int mi = 0; mi < 4; ++mi) {
      int m0 = blockIdx.y * 128 + wm + mi * 16 + half * 8;
      #pragma unroll
      for (int r = 0; r < 8; ++r) {
        float v = acc[mi][ni][r] + be;
        v = v > 0.f ? v : 0.f;
        __builtin_nontemporal_store(v, h + (size_t)(m0 + r) * HID_DIM + n);
      }
    }
  }
}

// ---------------- Kernel 2: exact top-32 radix select + mask + compact ----------------
__global__ __launch_bounds__(512)
void sae_topk_mask(float* __restrict__ h,
                   float* __restrict__ pairV,
                   int*   __restrict__ pairI)
{
  __shared__ unsigned int hist[256];
  __shared__ unsigned int scanBuf[512];

  const int b   = blockIdx.x;
  const int tid = threadIdx.x;
  float* row = h + (size_t)b * HID_DIM;

  // post-ReLU values are >= 0: bit pattern order == float order
  unsigned int bits[64];
  const uint4* rp = (const uint4*)(row + tid * 64);
  #pragma unroll
  for (int i = 0; i < 16; ++i) {
    uint4 q = rp[i];
    bits[4*i+0] = q.x; bits[4*i+1] = q.y; bits[4*i+2] = q.z; bits[4*i+3] = q.w;
  }

  unsigned int prefix = 0, pmask = 0, need = TOPK;
  for (int round = 0; round < 4; ++round) {
    int shift = 24 - 8 * round;
    if (tid < 256) hist[tid] = 0;
    __syncthreads();
    unsigned int pbits = prefix & pmask;
    #pragma unroll
    for (int j = 0; j < 64; ++j) {
      unsigned int v = bits[j];
      if ((v & pmask) == pbits) atomicAdd(&hist[(v >> shift) & 0xFFu], 1u);
    }
    __syncthreads();
    // all threads redundantly scan (uniform, LDS broadcast)
    unsigned int below = 0; int bsel = 0;
    for (int bin = 255; bin >= 0; --bin) {
      unsigned int c = hist[bin];
      if (below + c >= need) { bsel = bin; break; }
      below += c;
    }
    prefix |= (unsigned int)bsel << shift;
    pmask  |= 0xFFu << shift;
    need   -= below;
    __syncthreads();
  }

  const unsigned int tBits  = prefix;   // exact bits of 32nd-largest
  const unsigned int needEq = need;     // how many ==tBits to keep (>=1)
  const unsigned int nG     = TOPK - needEq;

  unsigned int cg = 0, ce = 0;
  #pragma unroll
  for (int j = 0; j < 64; ++j) {
    cg += (bits[j] >  tBits);
    ce += (bits[j] == tBits);
  }
  scanBuf[tid] = (cg << 16) | ce;
  __syncthreads();
  for (int off = 1; off < 512; off <<= 1) {
    unsigned int mine = scanBuf[tid];
    unsigned int add  = (tid >= off) ? scanBuf[tid - off] : 0u;
    __syncthreads();
    scanBuf[tid] = mine + add;
    __syncthreads();
  }
  unsigned int excl = scanBuf[tid] - ((cg << 16) | ce);
  unsigned int g = excl >> 16;
  unsigned int e = excl & 0xFFFFu;

  float* pv = pairV + b * TOPK;
  int*   pi = pairI + b * TOPK;
  #pragma unroll
  for (int j = 0; j < 64; ++j) {
    unsigned int v = bits[j];
    float fv = __uint_as_float(v);
    float outv = 0.f; int slot = -1;
    if (v > tBits)        { outv = fv; slot = (int)g; ++g; }
    else if (v == tBits)  { if (e < needEq) { outv = fv; slot = (int)(nG + e); } ++e; }
    __builtin_nontemporal_store(outv, row + tid * 64 + j);
    if (slot >= 0) { pv[slot] = fv; pi[slot] = tid * 64 + j; }
  }
}

// ---------------- Kernel 3: sparse decode (h @ W_dec^T + b_dec) ----------------
__global__ __launch_bounds__(256)
void sae_decode(const float* __restrict__ Wdec,
                const float* __restrict__ bDec,
                const float* __restrict__ pairV,
                const int*   __restrict__ pairI,
                float* __restrict__ xhat)
{
  __shared__ float sv[TOPK];
  __shared__ int   si[TOPK];
  const int b = blockIdx.x, tid = threadIdx.x;
  if (tid < TOPK) { sv[tid] = pairV[b * TOPK + tid]; si[tid] = pairI[b * TOPK + tid]; }
  __syncthreads();

  float acc0 = bDec[tid];
  float acc1 = bDec[tid + 256];
  float acc2 = bDec[tid + 512];
  float acc3 = bDec[tid + 768];
  #pragma unroll 4
  for (int p = 0; p < TOPK; ++p) {
    float v = sv[p];
    if (v != 0.f) {                // uniform across block: skip all-zero slots
      int id = si[p];
      acc0 += v * Wdec[(size_t)(tid      ) * HID_DIM + id];
      acc1 += v * Wdec[(size_t)(tid + 256) * HID_DIM + id];
      acc2 += v * Wdec[(size_t)(tid + 512) * HID_DIM + id];
      acc3 += v * Wdec[(size_t)(tid + 768) * HID_DIM + id];
    }
  }
  float* o = xhat + (size_t)b * IN_DIM;
  o[tid] = acc0; o[tid + 256] = acc1; o[tid + 512] = acc2; o[tid + 768] = acc3;
}

extern "C" void kernel_launch(void* const* d_in, const int* in_sizes, int n_in,
                              void* d_out, int out_size, void* d_ws, size_t ws_size,
                              hipStream_t stream)
{
  const float* x    = (const float*)d_in[0];
  const float* Wenc = (const float*)d_in[1];
  const float* bEnc = (const float*)d_in[2];
  const float* Wdec = (const float*)d_in[3];
  const float* bDec = (const float*)d_in[4];

  float* xhat = (float*)d_out;                       // [B, I]
  float* h    = xhat + (size_t)BATCH * IN_DIM;       // [B, H]

  float* pairV = (float*)d_ws;                                           // [B, 32]
  int*   pairI = (int*)((char*)d_ws + (size_t)BATCH * TOPK * sizeof(float)); // [B, 32]

  dim3 g1(HID_DIM / 128, BATCH / 128);
  sae_encode_wmma<<<g1, 256, 0, stream>>>(x, Wenc, bEnc, h);
  sae_topk_mask<<<BATCH, 512, 0, stream>>>(h, pairV, pairI);
  sae_decode<<<BATCH, 256, 0, stream>>>(Wdec, bDec, pairV, pairI, xhat);
}
